// SPGSiameseActor_70076686401854
// MI455X (gfx1250) — matrix operand order repaired and checked by reading, not compiled, not run on hardware
//
#include <hip/hip_runtime.h>
#include <hip/hip_bf16.h>
#include <math.h>

// ---------------------------------------------------------------------------
// SPG Siamese Actor forward for MI455X (gfx1250), wave32 + WMMA f16->f32.
// ---------------------------------------------------------------------------

typedef __attribute__((ext_vector_type(16))) _Float16 v16h;
typedef __attribute__((ext_vector_type(8)))  float    v8f;
typedef __attribute__((ext_vector_type(4)))  int      v4i;

union FragH { v16h v; _Float16 h[16]; uint4 q[2]; };
union AccF  { v8f  v; float    f[8]; };

// gfx1250 async global->LDS copy path (ASYNCcnt), guarded so compile never fails.
#if defined(__gfx1250__) && \
    __has_builtin(__builtin_amdgcn_global_load_async_to_lds_b128) && \
    __has_builtin(__builtin_amdgcn_s_wait_asynccnt)
#define USE_ASYNC_LDS 1
#else
#define USE_ASYNC_LDS 0
#endif

__device__ __forceinline__ void stage16(_Float16* dst_lds, const _Float16* src_global) {
#if USE_ASYNC_LDS
    // Signature (from clang diagnostic): (int4 AS1*, int4 AS3*, imm offset, imm cpol)
    __builtin_amdgcn_global_load_async_to_lds_b128(
        (__attribute__((address_space(1))) v4i*)src_global,
        (__attribute__((address_space(3))) v4i*)dst_lds,
        0, 0);
#else
    *(uint4*)dst_lds = *(const uint4*)src_global;
#endif
}

__device__ __forceinline__ void stage_wait() {
#if USE_ASYNC_LDS
    __builtin_amdgcn_s_wait_asynccnt(0);
#endif
}

// A-matrix fragment (16x32 f16, row-major source, stride ld halfs).
// Lane L<16: row L, K = {0..7, 16..23}; lane L>=16: row L-16, K = {8..15, 24..31}.
__device__ __forceinline__ v16h load_frag_a16(const _Float16* base, int ld) {
    int lane = threadIdx.x & 31;
    int row  = lane & 15;
    int hi   = lane >> 4;
    const _Float16* p = base + (size_t)row * ld + hi * 8;
    FragH a;
    a.q[0] = *(const uint4*)(p);        // K = kbase .. kbase+7
    a.q[1] = *(const uint4*)(p + 16);   // K = kbase+16 .. kbase+23
    return a.v;
}

// B-matrix fragment (32x16 f16, B[k][n] = W[n][k] with W row-major, stride ld).
// Lane L: column L&15; lanes 0-15 hold K={0..7,16..23}, lanes 16-31 hold K={8..15,24..31}.
__device__ __forceinline__ v16h load_frag_b16(const _Float16* base, int ld) {
    int lane = threadIdx.x & 31;
    int col  = lane & 15;
    int hi   = lane >> 4;
    const _Float16* p = base + (size_t)col * ld + hi * 8;
    FragH b;
    b.q[0] = *(const uint4*)(p);
    b.q[1] = *(const uint4*)(p + 16);
    return b.v;
}

// ---------------------------------------------------------------------------
// One-time conversion kernels
// ---------------------------------------------------------------------------
__global__ void cvt_f32_to_f16(const float* __restrict__ in, _Float16* __restrict__ out, int n) {
    int i = blockIdx.x * 256 + threadIdx.x;
    if (i < n) out[i] = (_Float16)in[i];
}

__global__ void add_bias_kernel(const float* __restrict__ a, const float* __restrict__ b,
                                float* __restrict__ out, int n) {
    int i = blockIdx.x * 256 + threadIdx.x;
    if (i < n) out[i] = a[i] + b[i];
}

// ---------------------------------------------------------------------------
// K1: G[b,n,e] = leaky_relu(x[b,n,:] @ W_emb[e,:] + b_emb[e]), f16 out.
// Rows r = b*512 + n (65536), N = 256, K = 32 (single WMMA). 1 wave / tile.
// ---------------------------------------------------------------------------
__global__ __launch_bounds__(128)
void embed_kernel(const float* __restrict__ x, const _Float16* __restrict__ Wh,
                  const float* __restrict__ bias, _Float16* __restrict__ G) {
    int wid  = threadIdx.x >> 5;
    int tile = blockIdx.x * 4 + wid;       // 65536 tiles = 4096 row-tiles x 16 col-tiles
    int rt = tile >> 4, ct = tile & 15;
    int r0 = rt * 16, e0 = ct * 16;

    int lane = threadIdx.x & 31;
    int row = lane & 15, hi = lane >> 4, kbase = hi * 8;

    // Pack f32 x row into f16 A fragment on the fly.
    const float* xr = x + (size_t)(r0 + row) * 32;
    FragH a;
#pragma unroll
    for (int j = 0; j < 8; ++j) {
        a.h[j]     = (_Float16)xr[kbase + j];
        a.h[j + 8] = (_Float16)xr[kbase + 16 + j];
    }
    v16h bf = load_frag_b16(Wh + (size_t)e0 * 32, 32);
    v8f c = {};
    c = __builtin_amdgcn_wmma_f32_16x16x32_f16(false, a.v, false, bf, (short)0, c, false, false);

    AccF acc; acc.v = c;
    int col = lane & 15, rb = hi * 8;
    float be = bias[e0 + col];
#pragma unroll
    for (int v = 0; v < 8; ++v) {
        float t = acc.f[v] + be;
        t = (t >= 0.f) ? t : 0.01f * t;                     // leaky_relu(0.01)
        G[(size_t)(r0 + rb + v) * 256 + (e0 + col)] = (_Float16)t;
    }
}

// ---------------------------------------------------------------------------
// K2: s[b,m,n] = sum_e g2[b,m,e] * g1[b,n,e]   (K=256 -> 8 WMMAs), f16 out.
// ---------------------------------------------------------------------------
__global__ __launch_bounds__(128)
void sim_kernel(const _Float16* __restrict__ G, _Float16* __restrict__ S) {
    int wid  = threadIdx.x >> 5;
    int tile = blockIdx.x * 4 + wid;       // 32768 tiles = 128 b x 16 mt x 16 nt
    int b  = tile >> 8;
    int mt = (tile >> 4) & 15, nt = tile & 15;
    int m0 = mt * 16, n0 = nt * 16;

    const _Float16* Abase = G + (size_t)(b * 512 + 256 + m0) * 256;  // g2 rows
    const _Float16* Bbase = G + (size_t)(b * 512 + n0) * 256;        // g1 rows as columns
    v8f c = {};
#pragma unroll
    for (int kk = 0; kk < 256; kk += 32) {
        v16h a  = load_frag_a16(Abase + kk, 256);
        v16h bb = load_frag_b16(Bbase + kk, 256);
        c = __builtin_amdgcn_wmma_f32_16x16x32_f16(false, a, false, bb, (short)0, c, false, false);
    }
    AccF acc; acc.v = c;
    int lane = threadIdx.x & 31, col = lane & 15, rb = (lane >> 4) * 8;
#pragma unroll
    for (int v = 0; v < 8; ++v)
        S[((size_t)b * 256 + m0 + rb + v) * 256 + n0 + col] = (_Float16)acc.f[v];
}

// ---------------------------------------------------------------------------
// K3: one LSTM time step. Block = (32 batch x 16 hidden) patch, 4 waves = 4 gates,
// two accumulator tiles per wave so each weight fragment feeds two WMMAs.
// gates = [s_t | h_{t-1}] (K=768) @ [W_ih | W_hh]^T + (b_ih+b_hh); fused cell.
// ---------------------------------------------------------------------------
__global__ __launch_bounds__(128)
void lstm_step_kernel(const _Float16* __restrict__ S,
                      const _Float16* __restrict__ Wih,   // 2048x256 f16
                      const _Float16* __restrict__ Whh,   // 2048x512 f16
                      const float* __restrict__ biasg,    // 2048 (b_ih+b_hh)
                      float* __restrict__ Cst,            // 128x512 cell state
                      _Float16* __restrict__ Hs,          // 128x256x512 hidden history
                      int t) {
    __shared__ __align__(16) _Float16 Alds[32 * 768];     // 48 KB shared A-panel
    __shared__ __align__(16) float    Gx[4][32][16];      // 8 KB gate exchange

    int bt = blockIdx.x;        // 0..3   (32-batch tile)
    int dt = blockIdx.y;        // 0..31  (hidden tile)
    int tid = threadIdx.x;

    // Stage A panel: 32 rows x [256 (s_t) | 512 (h_{t-1})] halfs, async 16B copies.
    for (int cid = tid; cid < 1024; cid += 128) {
        int r = cid >> 5, cc = cid & 31;
        int b = bt * 32 + r;
        stage16(&Alds[r * 768 + cc * 8], S + ((size_t)b * 256 + t) * 256 + cc * 8);
    }
    if (t > 0) {
        for (int cid = tid; cid < 2048; cid += 128) {
            int r = cid >> 6, cc = cid & 63;
            int b = bt * 32 + r;
            stage16(&Alds[r * 768 + 256 + cc * 8],
                    Hs + ((size_t)b * 256 + (t - 1)) * 512 + cc * 8);
        }
    }
    stage_wait();           // s_wait_asynccnt 0 (per-wave), then block barrier
    __syncthreads();

    int wid = tid >> 5;                 // gate index: 0=i 1=f 2=g 3=o
    int j0  = wid * 512 + dt * 16;      // gate output column base
    v8f c0 = {}, c1 = {};
    int kend = (t > 0) ? 768 : 256;     // h_{-1} == 0 -> skip W_hh part at t=0
    for (int kk = 0; kk < kend; kk += 32) {
        const _Float16* wb = (kk < 256) ? (Wih + (size_t)j0 * 256 + kk)
                                        : (Whh + (size_t)j0 * 512 + (kk - 256));
        v16h bb = load_frag_b16(wb, (kk < 256) ? 256 : 512);
        v16h a0 = load_frag_a16(&Alds[kk], 768);                // batches bt*32 .. +15
        v16h a1 = load_frag_a16(&Alds[16 * 768 + kk], 768);     // batches bt*32+16 .. +31
        c0 = __builtin_amdgcn_wmma_f32_16x16x32_f16(false, a0, false, bb, (short)0, c0, false, false);
        c1 = __builtin_amdgcn_wmma_f32_16x16x32_f16(false, a1, false, bb, (short)0, c1, false, false);
    }
    AccF acc0, acc1; acc0.v = c0; acc1.v = c1;
    int lane = tid & 31, col = lane & 15, rb = (lane >> 4) * 8;
    float bg = biasg[j0 + col];
#pragma unroll
    for (int v = 0; v < 8; ++v) {
        Gx[wid][rb + v][col]      = acc0.f[v] + bg;
        Gx[wid][16 + rb + v][col] = acc1.f[v] + bg;
    }
    __syncthreads();

    // Fused LSTM cell update (512 elements, 4 per thread).
    for (int idx = tid; idx < 512; idx += 128) {
        int r = idx >> 4, dc = idx & 15;
        float ig = Gx[0][r][dc], fg = Gx[1][r][dc], gg = Gx[2][r][dc], og = Gx[3][r][dc];
        ig = 1.f / (1.f + expf(-ig));
        fg = 1.f / (1.f + expf(-fg));
        og = 1.f / (1.f + expf(-og));
        gg = tanhf(gg);
        int b = bt * 32 + r, d = dt * 16 + dc;
        float cp = (t > 0) ? Cst[b * 512 + d] : 0.f;
        float cn = fg * cp + ig * gg;
        float hn = og * tanhf(cn);
        Cst[b * 512 + d] = cn;
        Hs[((size_t)b * 256 + t) * 512 + d] = (_Float16)hn;
    }
}

// ---------------------------------------------------------------------------
// K4: psi0 = exp(Hs @ W_fc^T + b_fc)  (rows 32768, N=256, K=512 -> 16 WMMAs)
// ---------------------------------------------------------------------------
__global__ __launch_bounds__(128)
void fc_exp_kernel(const _Float16* __restrict__ Hs, const _Float16* __restrict__ Wfc,
                   const float* __restrict__ bfc, float* __restrict__ out) {
    int wid  = threadIdx.x >> 5;
    int tile = blockIdx.x * 4 + wid;   // 32768 tiles = 2048 row-tiles x 16 col-tiles
    int rt = tile >> 4, ct = tile & 15;
    int r0 = rt * 16, j0 = ct * 16;
    v8f c = {};
#pragma unroll
    for (int kk = 0; kk < 512; kk += 32) {
        v16h a  = load_frag_a16(Hs + (size_t)r0 * 512 + kk, 512);
        v16h bb = load_frag_b16(Wfc + (size_t)j0 * 512 + kk, 512);
        c = __builtin_amdgcn_wmma_f32_16x16x32_f16(false, a, false, bb, (short)0, c, false, false);
    }
    AccF acc; acc.v = c;
    int lane = threadIdx.x & 31, col = lane & 15, rb = (lane >> 4) * 8;
    float be = bfc[j0 + col];
#pragma unroll
    for (int v = 0; v < 8; ++v)
        out[(size_t)(r0 + rb + v) * 256 + j0 + col] = expf(acc.f[v] + be);  // TAU = 1
}

// ---------------------------------------------------------------------------
// K5: all 5 Sinkhorn iterations fused; one block per batch; 256x256 matrix in
// LDS with +1 padding (256*257*4 B = 257 KB of the 320 KB WGP LDS).
// ---------------------------------------------------------------------------
__global__ __launch_bounds__(256)
void sinkhorn_kernel(float* __restrict__ psi) {
    extern __shared__ float P[];       // 256 * 257 floats
    int b   = blockIdx.x;
    int tid = threadIdx.x;             // 0..255
    float* base = psi + (size_t)b * 65536;

    for (int m = 0; m < 256; ++m)                   // coalesced load
        P[m * 257 + tid] = base[m * 256 + tid];
    __syncthreads();

    for (int it = 0; it < 5; ++it) {
        // normalize over axis 2 (n): thread owns row `tid`
        float s = 0.f;
        for (int n = 0; n < 256; ++n) s += P[tid * 257 + n];
        float inv = 1.f / s;
        for (int n = 0; n < 256; ++n) P[tid * 257 + n] *= inv;
        __syncthreads();
        // normalize over axis 1 (m): thread owns column `tid`
        s = 0.f;
        for (int m = 0; m < 256; ++m) s += P[m * 257 + tid];
        inv = 1.f / s;
        for (int m = 0; m < 256; ++m) P[m * 257 + tid] *= inv;
        __syncthreads();
    }
    for (int m = 0; m < 256; ++m)                   // coalesced store
        base[m * 256 + tid] = P[m * 257 + tid];
}

// ---------------------------------------------------------------------------
// Host launcher
// ---------------------------------------------------------------------------
extern "C" void kernel_launch(void* const* d_in, const int* in_sizes, int n_in,
                              void* d_out, int out_size, void* d_ws, size_t ws_size,
                              hipStream_t stream) {
    (void)in_sizes; (void)n_in; (void)out_size; (void)ws_size;

    const float* x    = (const float*)d_in[0];   // 128 x 512 x 32
    const float* Wemb = (const float*)d_in[1];   // 256 x 32
    const float* bemb = (const float*)d_in[2];   // 256
    const float* Wih  = (const float*)d_in[3];   // 2048 x 256
    const float* Whh  = (const float*)d_in[4];   // 2048 x 512
    const float* bih  = (const float*)d_in[5];   // 2048
    const float* bhh  = (const float*)d_in[6];   // 2048
    const float* Wfc  = (const float*)d_in[7];   // 256 x 512
    const float* bfc  = (const float*)d_in[8];   // 256
    float* out = (float*)d_out;                  // 128 x 256 x 256

    // Workspace carve-out (~88 MB total), 256 B aligned regions.
    char* ws = (char*)d_ws;
    size_t off = 0;
    auto carve = [&](size_t bytes) -> void* {
        void* p = ws + off;
        off = (off + bytes + 255) & ~(size_t)255;
        return p;
    };
    _Float16* Wemb_h = (_Float16*)carve(256u * 32 * 2);
    _Float16* Wih_h  = (_Float16*)carve(2048u * 256 * 2);
    _Float16* Whh_h  = (_Float16*)carve(2048u * 512 * 2);
    _Float16* Wfc_h  = (_Float16*)carve(256u * 512 * 2);
    float*    biasg  = (float*)   carve(2048u * 4);
    _Float16* G      = (_Float16*)carve((size_t)128 * 512 * 256 * 2);  // 33.5 MB
    _Float16* S      = (_Float16*)carve((size_t)128 * 256 * 256 * 2);  // 16.8 MB
    _Float16* Hs     = (_Float16*)carve((size_t)128 * 256 * 512 * 2);  // 33.5 MB
    float*    Cst    = (float*)   carve((size_t)128 * 512 * 4);        // 256 KB

    // One-time weight conversion (f32 -> f16, L2-resident afterwards).
    cvt_f32_to_f16<<<(8192 + 255) / 256, 256, 0, stream>>>(Wemb, Wemb_h, 8192);
    cvt_f32_to_f16<<<(524288 + 255) / 256, 256, 0, stream>>>(Wih, Wih_h, 524288);
    cvt_f32_to_f16<<<(1048576 + 255) / 256, 256, 0, stream>>>(Whh, Whh_h, 1048576);
    cvt_f32_to_f16<<<(131072 + 255) / 256, 256, 0, stream>>>(Wfc, Wfc_h, 131072);
    add_bias_kernel<<<8, 256, 0, stream>>>(bih, bhh, biasg, 2048);

    // Embedding + leaky ReLU: 65536x256 output, 65536 tiles / 4 waves per block.
    embed_kernel<<<16384, 128, 0, stream>>>(x, Wemb_h, bemb, G);

    // Batched similarity: 128 x (256x256x256), 32768 tiles.
    sim_kernel<<<8192, 128, 0, stream>>>(G, S);

    // Sequential LSTM: 256 stream-ordered steps (graph-capturable).
    for (int t = 0; t < 256; ++t)
        lstm_step_kernel<<<dim3(4, 32), 128, 0, stream>>>(S, Wih_h, Whh_h, biasg, Cst, Hs, t);

    // FC + exp into d_out, then fully-fused 5-iteration Sinkhorn.
    fc_exp_kernel<<<8192, 128, 0, stream>>>(Hs, Wfc_h, bfc, out);
    sinkhorn_kernel<<<128, 256, 256 * 257 * sizeof(float), stream>>>(out);
}
